// LSTM_GAT_575525617900
// MI455X (gfx1250) — matrix operand order repaired and checked by reading, not compiled, not run on hardware
//
#include <hip/hip_runtime.h>
#include <math.h>

#define LH    64      // LSTM hidden
#define G4    256     // 4*LH gate width
#define TT    24      // time steps
#define NSTAT 8       // static features
#define XW    32      // x row width = NSTAT + TT
#define HC    128     // H*C GAT width
#define CK    96      // padded combined width (64+8 -> 96, multiple of 32)

typedef __attribute__((ext_vector_type(16))) __bf16 bf16x16;
typedef __attribute__((ext_vector_type(8)))  __bf16 bf16x8;
typedef __attribute__((ext_vector_type(8)))  float  f32x8;

static __device__ __forceinline__ f32x8 wmma_bf16(bf16x16 a, bf16x16 b, f32x8 c) {
  return __builtin_amdgcn_wmma_f32_16x16x32_bf16(false, a, false, b, (short)0, c,
                                                 false, false);
}
// branch-free sigmoid: v_exp_f32 + v_rcp_f32 (no precise-div expansion)
static __device__ __forceinline__ float sigmoidf_(float v) {
  return __builtin_amdgcn_rcpf(1.0f + __expf(-v));
}
// branch-free tanh via sigmoid identity (libm tanhf has divergent branches)
static __device__ __forceinline__ float tanhf_(float v) {
  return 2.0f * sigmoidf_(2.0f * v) - 1.0f;
}

// A operand (16x32 bf16) from a row-major [16][LH] bf16 tile in LDS.
// A layout: lanes 0-15 = rows M, K={0..7,16..23}; lanes 16-31 same rows, K={8..15,24..31}.
static __device__ __forceinline__ bf16x16 load_A64(const __bf16* hs, int k0, int lane) {
  const int m  = lane & 15;
  const int kh = (lane >> 4) << 3;  // 0 or 8
  const bf16x8 lo = *(const bf16x8*)(hs + m * LH + k0 + kh);
  const bf16x8 hi = *(const bf16x8*)(hs + m * LH + k0 + 16 + kh);
  return __builtin_shufflevector(lo, hi, 0, 1, 2, 3, 4, 5, 6, 7,
                                 8, 9, 10, 11, 12, 13, 14, 15);
}

// B operand (32x16 bf16) from row-major [G][K] weight in LDS (K contiguous per row).
// B layout: lanes 0-15 = col N, K={0..7,16..23}; lanes 16-31 same cols, K={8..15,24..31}.
// `opq` is always 0 but is laundered through asm volatile by the caller each
// subtile, so these LDS loads are NOT loop-invariant and cannot be hoisted
// (round-1 showed LICM hoisting ~768 VGPRs of weights and spilling to scratch).
static __device__ __forceinline__ bf16x16 load_Bk(const __bf16* w, int ldk, int nt,
                                                  int k0, int lane, int opq) {
  const int g  = nt * 16 + (lane & 15);
  const int kb = k0 + ((lane >> 4) << 3);  // 0 or 8
  const bf16x8* p = (const bf16x8*)(w + opq + g * ldk + kb);
  return __builtin_shufflevector(p[0], p[2], 0, 1, 2, 3, 4, 5, 6, 7,
                                 8, 9, 10, 11, 12, 13, 14, 15);
}

// ---------------------------------------------------------------------------
// Fused 2-layer LSTM: one wave owns 16 nodes for all 24 steps. Weights in LDS
// (bf16), c-state in VGPRs (f32, WMMA C-layout), h staged via LDS tiles so the
// C-layout result can be re-read in A-layout. Per 16-column subtile we run all
// four gates to keep live registers small.
// combined = [h2_last(64) | static(8) | 0-pad(24)] in bf16.
// ---------------------------------------------------------------------------
__global__ __launch_bounds__(256) void lstm_fused_kernel(
    const float* __restrict__ x,
    const float* __restrict__ w_ih0, const float* __restrict__ w_hh0,
    const float* __restrict__ b_ih0, const float* __restrict__ b_hh0,
    const float* __restrict__ w_ih1, const float* __restrict__ w_hh1,
    const float* __restrict__ b_ih1, const float* __restrict__ b_hh1,
    __bf16* __restrict__ combined, int n)
{
  __shared__ __align__(16) __bf16 wh0[G4 * LH];
  __shared__ __align__(16) __bf16 wi1[G4 * LH];
  __shared__ __align__(16) __bf16 wh1[G4 * LH];
  __shared__ float bias0[G4], bias1[G4], wx0[G4];
  __shared__ __align__(16) __bf16 hstage[8][2][16 * LH];
  __shared__ float xdyn[8][16][TT];

  const int tid = threadIdx.x;
  for (int i = tid; i < G4 * LH; i += 256) {
    wh0[i] = (__bf16)w_hh0[i];
    wi1[i] = (__bf16)w_ih1[i];
    wh1[i] = (__bf16)w_hh1[i];
  }
  for (int i = tid; i < G4; i += 256) {
    bias0[i] = b_ih0[i] + b_hh0[i];
    bias1[i] = b_ih1[i] + b_hh1[i];
    wx0[i]   = w_ih0[i];
  }
  const int wave = tid >> 5;
  const int lane = tid & 31;
  const int nodeBase = blockIdx.x * 128 + wave * 16;
  for (int i = lane; i < 16 * TT; i += 32) {
    int nn = i / TT, t = i % TT;
    int gn = nodeBase + nn;
    xdyn[wave][nn][t] = (gn < n) ? x[(size_t)gn * XW + NSTAT + t] : 0.0f;
  }
  __bf16* hs0 = &hstage[wave][0][0];
  __bf16* hs1 = &hstage[wave][1][0];
  for (int i = lane; i < 16 * LH; i += 32) {
    hs0[i] = (__bf16)0.0f;
    hs1[i] = (__bf16)0.0f;
  }
  __syncthreads();

  const int mb  = (lane >> 4) << 3;  // 0 or 8: row base within C-tile
  const int col = lane & 15;

  f32x8 c0[4], c1[4];
#pragma unroll
  for (int s = 0; s < 4; ++s)
#pragma unroll
    for (int v = 0; v < 8; ++v) { c0[s][v] = 0.0f; c1[s][v] = 0.0f; }

  int opq = 0;  // laundered zero: defeats LICM on weight B-tile loads

  // one 16x16 gate tile for layer 0:  bias + x*w_ih + h0 @ w_hh0
  auto tile0 = [&](int nt, const float* xv, bf16x16 a0, bf16x16 a1) -> f32x8 {
    const int g = nt * 16 + col;
    const float bb = bias0[g];
    const float wx = wx0[g];
    f32x8 acc;
#pragma unroll
    for (int v = 0; v < 8; ++v) acc[v] = bb + xv[v] * wx;
    acc = wmma_bf16(a0, load_Bk(wh0, LH, nt, 0,  lane, opq), acc);
    acc = wmma_bf16(a1, load_Bk(wh0, LH, nt, 32, lane, opq), acc);
    return acc;
  };
  // one 16x16 gate tile for layer 1:  bias + h1in @ w_ih1 + h1prev @ w_hh1
  auto tile1 = [&](int nt, bf16x16 ai0, bf16x16 ai1, bf16x16 ah0,
                   bf16x16 ah1) -> f32x8 {
    const int g = nt * 16 + col;
    const float bb = bias1[g];
    f32x8 acc;
#pragma unroll
    for (int v = 0; v < 8; ++v) acc[v] = bb;
    acc = wmma_bf16(ai0, load_Bk(wi1, LH, nt, 0,  lane, opq), acc);
    acc = wmma_bf16(ai1, load_Bk(wi1, LH, nt, 32, lane, opq), acc);
    acc = wmma_bf16(ah0, load_Bk(wh1, LH, nt, 0,  lane, opq), acc);
    acc = wmma_bf16(ah1, load_Bk(wh1, LH, nt, 32, lane, opq), acc);
    return acc;
  };

#pragma unroll 1
  for (int t = 0; t < TT; ++t) {
    // ---------------- layer 0 ----------------
    {
      bf16x16 a0 = load_A64(hs0, 0, lane);
      bf16x16 a1 = load_A64(hs0, 32, lane);
      float xv[8];
#pragma unroll
      for (int v = 0; v < 8; ++v) xv[v] = xdyn[wave][mb + v][t];
#pragma unroll
      for (int s = 0; s < 4; ++s) {
        asm volatile("" : "+v"(opq));       // re-launder per subtile
        f32x8 ig = tile0(s,      xv, a0, a1);   // input gate  (cols   0..63)
        f32x8 gg = tile0(8 + s,  xv, a0, a1);   // cell gate   (cols 128..191)
        f32x8 fg = tile0(4 + s,  xv, a0, a1);   // forget gate (cols  64..127)
        f32x8 og = tile0(12 + s, xv, a0, a1);   // output gate (cols 192..255)
#pragma unroll
        for (int v = 0; v < 8; ++v) {
          float cv = sigmoidf_(fg[v]) * c0[s][v] + sigmoidf_(ig[v]) * tanhf_(gg[v]);
          c0[s][v] = cv;
          float hv = sigmoidf_(og[v]) * tanhf_(cv);
          hs0[(mb + v) * LH + s * 16 + col] = (__bf16)hv;
        }
      }
    }
    // ---------------- layer 1 ----------------
    {
      bf16x16 ai0 = load_A64(hs0, 0, lane);
      bf16x16 ai1 = load_A64(hs0, 32, lane);
      bf16x16 ah0 = load_A64(hs1, 0, lane);
      bf16x16 ah1 = load_A64(hs1, 32, lane);
#pragma unroll
      for (int s = 0; s < 4; ++s) {
        asm volatile("" : "+v"(opq));       // re-launder per subtile
        f32x8 ig = tile1(s,      ai0, ai1, ah0, ah1);
        f32x8 gg = tile1(8 + s,  ai0, ai1, ah0, ah1);
        f32x8 fg = tile1(4 + s,  ai0, ai1, ah0, ah1);
        f32x8 og = tile1(12 + s, ai0, ai1, ah0, ah1);
#pragma unroll
        for (int v = 0; v < 8; ++v) {
          float cv = sigmoidf_(fg[v]) * c1[s][v] + sigmoidf_(ig[v]) * tanhf_(gg[v]);
          c1[s][v] = cv;
          float hv = sigmoidf_(og[v]) * tanhf_(cv);
          hs1[(mb + v) * LH + s * 16 + col] = (__bf16)hv;
        }
      }
    }
  }

  // combined = [h2_last | static | pad]
  for (int i = lane; i < 16 * LH; i += 32) {
    int nn = i >> 6, k = i & 63;
    int gn = nodeBase + nn;
    if (gn < n) combined[(size_t)gn * CK + k] = hs1[i];
  }
  for (int i = lane; i < 16 * 32; i += 32) {
    int nn = i >> 5, cc = i & 31;
    int gn = nodeBase + nn;
    if (gn < n)
      combined[(size_t)gn * CK + LH + cc] =
          (__bf16)((cc < NSTAT) ? x[(size_t)gn * XW + cc] : 0.0f);
  }
}

// ---------------------------------------------------------------------------
// xl = combined(N x 96,bf16) @ lin_w^T (pad K 72->96), f32 out, via WMMA.
// ---------------------------------------------------------------------------
__global__ __launch_bounds__(256) void gat_lin_kernel(
    const __bf16* __restrict__ combined, const float* __restrict__ lin_w,
    float* __restrict__ xl, int n)
{
  __shared__ __align__(16) __bf16 wl[HC * CK];
  const int tid = threadIdx.x;
  for (int i = tid; i < HC * CK; i += 256) {
    int g = i / CK, k = i % CK;
    wl[i] = (__bf16)((k < 72) ? lin_w[g * 72 + k] : 0.0f);
  }
  __syncthreads();
  const int wave = tid >> 5, lane = tid & 31;
  const int nodeBase = blockIdx.x * 128 + wave * 16;
  const int m = lane & 15, kh = (lane >> 4) << 3, mb = (lane >> 4) << 3;
  const int col = lane & 15;

  int row = nodeBase + m;
  if (row >= n) row = n - 1;
  const __bf16* rp = combined + (size_t)row * CK;
  bf16x16 a[3];
#pragma unroll
  for (int ks = 0; ks < 3; ++ks) {
    const int k0 = ks * 32;
    const bf16x8 lo = *(const bf16x8*)(rp + k0 + kh);
    const bf16x8 hi = *(const bf16x8*)(rp + k0 + 16 + kh);
    a[ks] = __builtin_shufflevector(lo, hi, 0, 1, 2, 3, 4, 5, 6, 7,
                                    8, 9, 10, 11, 12, 13, 14, 15);
  }
  int opq = 0;
#pragma unroll
  for (int nt = 0; nt < 8; ++nt) {
    asm volatile("" : "+v"(opq));  // keep B loads interleaved, not bulk-hoisted
    f32x8 acc;
#pragma unroll
    for (int v = 0; v < 8; ++v) acc[v] = 0.0f;
#pragma unroll
    for (int ks = 0; ks < 3; ++ks)
      acc = wmma_bf16(a[ks], load_Bk(wl, CK, nt, ks * 32, lane, opq), acc);
#pragma unroll
    for (int v = 0; v < 8; ++v) {
      int gn = nodeBase + mb + v;
      if (gn < n) xl[(size_t)gn * HC + nt * 16 + col] = acc[v];
    }
  }
}

// ---------------------------------------------------------------------------
// GAT scalar stages
// ---------------------------------------------------------------------------
__global__ void gat_init_kernel(float* mmax, float* denom, float* gout,
                                float* scal, int n) {
  int idx = blockIdx.x * blockDim.x + threadIdx.x;
  if (idx < n * 4) { mmax[idx] = -INFINITY; denom[idx] = 0.0f; }
  if (idx < n * HC) gout[idx] = 0.0f;
  if (idx < 8) scal[idx] = 0.0f;
}

__global__ void gat_att_kernel(const float* __restrict__ xl,
                               const float* __restrict__ att_src,
                               const float* __restrict__ att_dst,
                               float* __restrict__ asrc, float* __restrict__ adst,
                               int n) {
  int idx = blockIdx.x * blockDim.x + threadIdx.x;
  if (idx >= n * 4) return;
  int nn = idx >> 2, h = idx & 3;
  float ss = 0.0f, dd = 0.0f;
  const float* xp = xl + (size_t)nn * HC + h * 32;
#pragma unroll
  for (int c = 0; c < 32; ++c) {
    float v = xp[c];
    ss += v * att_src[h * 32 + c];
    dd += v * att_dst[h * 32 + c];
  }
  asrc[idx] = ss;
  adst[idx] = dd;
}

__global__ void ea_reduce_kernel(const float* __restrict__ ea, float* scal, int E) {
  int idx = blockIdx.x * blockDim.x + threadIdx.x;
  float p = 0.0f;
  for (int i = idx; i < E; i += gridDim.x * blockDim.x) p += ea[i];
  atomicAdd(&scal[5], p);
}

__global__ void gat_prep_kernel(const float* __restrict__ lew,
                                const float* __restrict__ ae, float* scal, int E) {
  if (threadIdx.x == 0 && blockIdx.x == 0) {
    scal[0] = scal[5] / (float)E;
#pragma unroll
    for (int h = 0; h < 4; ++h) {
      float s = 0.0f;
      for (int c = 0; c < 32; ++c) s += lew[h * 32 + c] * ae[h * 32 + c];
      scal[1 + h] = s;
    }
  }
}

static __device__ __forceinline__ void atomic_max_f32(float* addr, float v) {
  if (v >= 0.0f)
    atomicMax((int*)addr, __float_as_int(v));
  else
    atomicMin((unsigned int*)addr, (unsigned int)__float_as_int(v));
}

__global__ void gat_max_kernel(const long long* __restrict__ ei,
                               const float* __restrict__ ea,
                               const float* __restrict__ asrc,
                               const float* __restrict__ adst,
                               const float* __restrict__ scal,
                               float* __restrict__ mmax, int E, int n) {
  int e = blockIdx.x * blockDim.x + threadIdx.x;
  if (e >= E + n) return;
  long long s, d; float w;
  if (e < E) { s = ei[e]; d = ei[E + e]; w = ea[e]; }
  else       { s = d = e - E;            w = scal[0]; }
#pragma unroll
  for (int h = 0; h < 4; ++h) {
    float l = asrc[s * 4 + h] + adst[d * 4 + h] + w * scal[1 + h];
    l = (l > 0.0f) ? l : 0.2f * l;
    atomic_max_f32(&mmax[d * 4 + h], l);
  }
}

__global__ void gat_ex_kernel(const long long* __restrict__ ei,
                              const float* __restrict__ ea,
                              const float* __restrict__ asrc,
                              const float* __restrict__ adst,
                              const float* __restrict__ scal,
                              const float* __restrict__ mmax,
                              float* __restrict__ exbuf, float* __restrict__ denom,
                              int E, int n) {
  int e = blockIdx.x * blockDim.x + threadIdx.x;
  if (e >= E + n) return;
  long long s, d; float w;
  if (e < E) { s = ei[e]; d = ei[E + e]; w = ea[e]; }
  else       { s = d = e - E;            w = scal[0]; }
#pragma unroll
  for (int h = 0; h < 4; ++h) {
    float l = asrc[s * 4 + h] + adst[d * 4 + h] + w * scal[1 + h];
    l = (l > 0.0f) ? l : 0.2f * l;
    float ex = __expf(l - mmax[d * 4 + h]);
    exbuf[(size_t)e * 4 + h] = ex;
    atomicAdd(&denom[d * 4 + h], ex);
  }
}

__global__ void gat_msg_kernel(const long long* __restrict__ ei,
                               const float* __restrict__ xl,
                               const float* __restrict__ exbuf,
                               const float* __restrict__ denom,
                               float* __restrict__ gout, int E, int n) {
  int idx = blockIdx.x * blockDim.x + threadIdx.x;
  if (idx >= (E + n) * 4) return;
  int e = idx >> 2, h = idx & 3;
  long long s, d;
  if (e < E) { s = ei[e]; d = ei[E + e]; }
  else       { s = d = e - E; }
  float alpha = exbuf[idx] * __builtin_amdgcn_rcpf(denom[d * 4 + h] + 1e-16f);
  const float* xp = xl + (size_t)s * HC + h * 32;
  float* op = gout + (size_t)d * HC + h * 32;
  __builtin_prefetch(xp, 0, 1);
#pragma unroll
  for (int c = 0; c < 32; ++c) atomicAdd(&op[c], alpha * xp[c]);
}

__global__ void gat_final_kernel(const float* __restrict__ gout,
                                 const float* __restrict__ gat_bias,
                                 const float* __restrict__ fc_w,
                                 const float* __restrict__ fc_b,
                                 float* __restrict__ out, int n) {
  int nn = blockIdx.x * blockDim.x + threadIdx.x;
  if (nn >= n) return;
  float acc[4] = {0.0f, 0.0f, 0.0f, 0.0f};
  const float* gp = gout + (size_t)nn * HC;
  for (int j = 0; j < HC; ++j) {
    float v = gp[j] + gat_bias[j];
    v = (v > 0.0f) ? v : (__expf(v) - 1.0f);
#pragma unroll
    for (int o = 0; o < 4; ++o) acc[o] += v * fc_w[o * HC + j];
  }
#pragma unroll
  for (int o = 0; o < 4; ++o) out[(size_t)nn * 4 + o] = fmaxf(acc[o] + fc_b[o], 0.0f);
}

// ---------------------------------------------------------------------------
extern "C" void kernel_launch(void* const* d_in, const int* in_sizes, int n_in,
                              void* d_out, int out_size, void* d_ws, size_t ws_size,
                              hipStream_t stream) {
  const float*     x        = (const float*)d_in[0];
  const long long* ei       = (const long long*)d_in[1];
  const float*     ea       = (const float*)d_in[2];
  const float*     w_ih0    = (const float*)d_in[3];
  const float*     w_hh0    = (const float*)d_in[4];
  const float*     b_ih0    = (const float*)d_in[5];
  const float*     b_hh0    = (const float*)d_in[6];
  const float*     w_ih1    = (const float*)d_in[7];
  const float*     w_hh1    = (const float*)d_in[8];
  const float*     b_ih1    = (const float*)d_in[9];
  const float*     b_hh1    = (const float*)d_in[10];
  const float*     lin_w    = (const float*)d_in[11];
  const float*     lin_ew   = (const float*)d_in[12];
  const float*     att_src  = (const float*)d_in[13];
  const float*     att_dst  = (const float*)d_in[14];
  const float*     att_edge = (const float*)d_in[15];
  const float*     gat_bias = (const float*)d_in[16];
  const float*     fc_w     = (const float*)d_in[17];
  const float*     fc_b     = (const float*)d_in[18];
  float* out = (float*)d_out;
  (void)n_in; (void)out_size; (void)ws_size;

  const int N = in_sizes[0] / XW;
  const int E = in_sizes[1] / 2;

  char*  ws  = (char*)d_ws;
  size_t off = 0;
  auto take = [&](size_t b) -> void* {
    void* p = ws + off;
    off += (b + 255) & ~(size_t)255;
    return p;
  };
  __bf16* combined = (__bf16*)take((size_t)N * CK * sizeof(__bf16));
  float*  xl       = (float*)take((size_t)N * HC * sizeof(float));
  float*  asrc     = (float*)take((size_t)N * 4 * sizeof(float));
  float*  adst     = (float*)take((size_t)N * 4 * sizeof(float));
  float*  mmax     = (float*)take((size_t)N * 4 * sizeof(float));
  float*  denom    = (float*)take((size_t)N * 4 * sizeof(float));
  float*  gout     = (float*)take((size_t)N * HC * sizeof(float));
  float*  exbuf    = (float*)take((size_t)(E + N) * 4 * sizeof(float));
  float*  scal     = (float*)take(256);

  const int tpb = 256;
  gat_init_kernel<<<(N * HC + tpb - 1) / tpb, tpb, 0, stream>>>(mmax, denom, gout, scal, N);
  lstm_fused_kernel<<<(N + 127) / 128, tpb, 0, stream>>>(
      x, w_ih0, w_hh0, b_ih0, b_hh0, w_ih1, w_hh1, b_ih1, b_hh1, combined, N);
  gat_lin_kernel<<<(N + 127) / 128, tpb, 0, stream>>>(combined, lin_w, xl, N);
  gat_att_kernel<<<(N * 4 + tpb - 1) / tpb, tpb, 0, stream>>>(xl, att_src, att_dst,
                                                              asrc, adst, N);
  ea_reduce_kernel<<<512, tpb, 0, stream>>>(ea, scal, E);
  gat_prep_kernel<<<1, 64, 0, stream>>>(lin_ew, att_edge, scal, E);
  gat_max_kernel<<<(E + N + tpb - 1) / tpb, tpb, 0, stream>>>(ei, ea, asrc, adst,
                                                              scal, mmax, E, N);
  gat_ex_kernel<<<(E + N + tpb - 1) / tpb, tpb, 0, stream>>>(ei, ea, asrc, adst,
                                                             scal, mmax, exbuf, denom, E, N);
  gat_msg_kernel<<<((E + N) * 4 + tpb - 1) / tpb, tpb, 0, stream>>>(ei, xl, exbuf,
                                                                    denom, gout, E, N);
  gat_final_kernel<<<(N + tpb - 1) / tpb, tpb, 0, stream>>>(gout, gat_bias, fc_w,
                                                            fc_b, out, N);
}
